// KANEX_10831907521101
// MI455X (gfx1250) — matrix operand order repaired and checked by reading, not compiled, not run on hardware
//
#include <hip/hip_runtime.h>
#include <hip/hip_bf16.h>
#include <math.h>

// ---------------------------------------------------------------------------
// KANEX transformer forward for MI455X (gfx1250, wave32, WMMA bf16->f32).
// B=2, N=1024, V=32000, D=1024, H=16, HD=32, DF=4096, L=2, WIN=16
// ---------------------------------------------------------------------------

typedef __bf16 bf16_t;
typedef float  v8f   __attribute__((ext_vector_type(8)));
typedef bf16_t v16bf __attribute__((ext_vector_type(16)));
typedef bf16_t v8bf  __attribute__((ext_vector_type(8)));

#define TM 128
#define TN 128
#define TK 32

// ---------------------------------------------------------------------------
// Generic GEMM: C[M,N] = act(A[M,K] @ W[K,N] + bias[N])
// A, W are f32 in HBM; converted to bf16 while staging into LDS.
// Double-buffered LDS (2 x 16KB), register-staged prefetch of tile k+1
// issued before the WMMAs of tile k; one barrier per K-step.
// 256 threads = 8 waves; wave (wr,wc) computes a 32x64 subtile via
// 2x4 grid of v_wmma_f32_16x16x32_bf16. Requires M%128==0, N%128==0, K%32==0.
// act: 0 = identity, 1 = SiLU
// ---------------------------------------------------------------------------
__global__ __launch_bounds__(256) void gemm_bias_act_kernel(
    const float* __restrict__ A, const float* __restrict__ W,
    const float* __restrict__ bias, float* __restrict__ C,
    int M, int N, int K, int act)
{
    __shared__ __attribute__((aligned(16))) bf16_t As[2][TM * TK];  // [row][k]
    __shared__ __attribute__((aligned(16))) bf16_t Bs[2][TN * TK];  // [col][k] (transposed)

    const int tid  = threadIdx.x;
    const int bm   = blockIdx.y * TM;
    const int bn   = blockIdx.x * TN;
    const int wid  = tid >> 5;
    const int lane = tid & 31;
    const int half = lane >> 4;   // 0: lanes 0-15, 1: lanes 16-31
    const int l16  = lane & 15;
    const int wr   = wid & 3;     // 32-row slice within the 128-row tile
    const int wc   = wid >> 2;    // 64-col slice within the 128-col tile

    v8f acc[2][4] = {};

    // Register staging for the global->LDS pipeline (4 float4 per matrix).
    float4 ra[4], rb[4];

    auto load_tiles = [&](int k0) {
        #pragma unroll
        for (int i = 0; i < 4; ++i) {
            int j = tid + i * 256;                 // 0..1023
            ra[i] = *(const float4*)(A + (size_t)(bm + (j >> 3)) * K + k0 + ((j & 7) << 2));
            rb[i] = *(const float4*)(W + (size_t)(k0 + (j >> 5)) * N + bn + ((j & 31) << 2));
        }
    };
    auto store_tiles = [&](int buf) {
        #pragma unroll
        for (int i = 0; i < 4; ++i) {
            int j = tid + i * 256;
            bf16_t* da = &As[buf][(j >> 3) * TK + ((j & 7) << 2)];
            da[0] = (bf16_t)ra[i].x; da[1] = (bf16_t)ra[i].y;
            da[2] = (bf16_t)ra[i].z; da[3] = (bf16_t)ra[i].w;
            int kr = j >> 5;
            int c4 = (j & 31) << 2;
            Bs[buf][(c4 + 0) * TK + kr] = (bf16_t)rb[i].x;
            Bs[buf][(c4 + 1) * TK + kr] = (bf16_t)rb[i].y;
            Bs[buf][(c4 + 2) * TK + kr] = (bf16_t)rb[i].z;
            Bs[buf][(c4 + 3) * TK + kr] = (bf16_t)rb[i].w;
        }
    };

    const int nK = K / TK;
    load_tiles(0);
    store_tiles(0);
    __syncthreads();

    for (int kt = 0; kt < nK; ++kt) {
        const int cur = kt & 1;

        // Issue global loads for the next tile before touching LDS/WMMA.
        if (kt + 1 < nK) load_tiles((kt + 1) * TK);

        // A fragment (16x32 bf16), CDNA5 ISA 7.12.2 layout:
        //   elems 0..7  = A[row][half*8 + 0..7]
        //   elems 8..15 = A[row][16 + half*8 + 0..7]
        v16bf afrag[2];
        #pragma unroll
        for (int mi = 0; mi < 2; ++mi) {
            int row = wr * 32 + mi * 16 + l16;
            v8bf lo = *(const v8bf*)&As[cur][row * TK + half * 8];
            v8bf hi = *(const v8bf*)&As[cur][row * TK + 16 + half * 8];
            afrag[mi] = __builtin_shufflevector(lo, hi,
                0,1,2,3,4,5,6,7,8,9,10,11,12,13,14,15);
        }
        // B fragment (32x16 bf16): elems 0..7 = B[half*8 + 0..7][col],
        // elems 8..15 = B[16 + half*8 + 0..7][col]; Bs is [col][k].
        v16bf bfrag[4];
        #pragma unroll
        for (int ni = 0; ni < 4; ++ni) {
            int col = wc * 64 + ni * 16 + l16;
            v8bf lo = *(const v8bf*)&Bs[cur][col * TK + half * 8];
            v8bf hi = *(const v8bf*)&Bs[cur][col * TK + 16 + half * 8];
            bfrag[ni] = __builtin_shufflevector(lo, hi,
                0,1,2,3,4,5,6,7,8,9,10,11,12,13,14,15);
        }

        #pragma unroll
        for (int mi = 0; mi < 2; ++mi)
            #pragma unroll
            for (int ni = 0; ni < 4; ++ni)
                acc[mi][ni] = __builtin_amdgcn_wmma_f32_16x16x32_bf16(
                    /*neg_a=*/false, afrag[mi],
                    /*neg_b=*/false, bfrag[ni],
                    /*c_mod=*/(short)0, acc[mi][ni],
                    /*reuse_a=*/false, /*reuse_b=*/false);

        // Stage next tile into the other buffer (safe: last read of that
        // buffer was fenced by the barrier at the end of iteration kt-1).
        if (kt + 1 < nK) store_tiles((kt + 1) & 1);
        __syncthreads();
    }

    // Epilogue: C/D layout — VGPR r holds M = r + half*8, N = l16.
    #pragma unroll
    for (int mi = 0; mi < 2; ++mi) {
        #pragma unroll
        for (int ni = 0; ni < 4; ++ni) {
            int col  = bn + wc * 64 + ni * 16 + l16;
            float bv = bias[col];
            #pragma unroll
            for (int r = 0; r < 8; ++r) {
                int row = bm + wr * 32 + mi * 16 + half * 8 + r;
                float v = acc[mi][ni][r] + bv;
                if (act == 1)   // SiLU via v_rcp_f32 (no IEEE divide chain)
                    v = v * __builtin_amdgcn_rcpf(1.0f + __expf(-v));
                C[(size_t)row * N + col] = v;
            }
        }
    }
}

// ---------------------------------------------------------------------------
// Embedding + positional term: h[b,n,:] = emb[x[b,n]][:] + pos[n][:]
// float4-vectorized (D % 4 == 0).
// ---------------------------------------------------------------------------
__global__ __launch_bounds__(256) void embed_kernel(
    const int* __restrict__ x, const float* __restrict__ emb,
    const float* __restrict__ pos, float* __restrict__ h,
    int Ntok, int Nn, int Dd)
{
    int d4c = Dd >> 2;
    size_t idx = (size_t)blockIdx.x * blockDim.x + threadIdx.x;
    size_t tot = (size_t)Ntok * d4c;
    if (idx >= tot) return;
    int d4    = (int)(idx % d4c);
    size_t bt = idx / d4c;
    int n     = (int)(bt % Nn);
    float4 e = *(const float4*)(emb + (size_t)x[bt] * Dd + d4 * 4);
    float4 p = *(const float4*)(pos + (size_t)n * Dd + d4 * 4);
    float4 r; r.x = e.x + p.x; r.y = e.y + p.y; r.z = e.z + p.z; r.w = e.w + p.w;
    *(float4*)(h + bt * Dd + d4 * 4) = r;
}

// ---------------------------------------------------------------------------
// Windowed attention, online softmax. One wave32 per (b, h, query i).
// HD == 32 == wave width: lane <-> head dim. Mask: j <= i + WIN - 1.
// qkv layout: [b*N + n][3*H*HD], q at c, k at H*HD + c, v at 2*H*HD + c.
// ---------------------------------------------------------------------------
__global__ __launch_bounds__(256) void attn_kernel(
    const float* __restrict__ qkv, float* __restrict__ out,
    int Bn, int Nn, int Hh, int HDd, int win)
{
    const int wid  = threadIdx.x >> 5;
    const int lane = threadIdx.x & 31;
    long gw    = (long)blockIdx.x * 8 + wid;
    long total = (long)Bn * Hh * Nn;
    if (gw >= total) return;
    int i = (int)(gw % Nn);
    long t = gw / Nn;
    int h  = (int)(t % Hh);
    int b  = (int)(t / Hh);

    const int   C     = 3 * Hh * HDd;
    const float scale = rsqrtf((float)HDd);
    const float qd    = qkv[((size_t)(b * Nn + i)) * C + h * HDd + lane];

    float m = -3.0e30f, s = 0.0f, o = 0.0f;
    int jmax = i + win - 1;
    if (jmax > Nn - 1) jmax = Nn - 1;

    for (int j = 0; j <= jmax; ++j) {
        const float* kp = qkv + ((size_t)(b * Nn + j)) * C + Hh * HDd + h * HDd;
        const float* vp = kp + Hh * HDd;
        float part = qd * kp[lane];
        #pragma unroll
        for (int off = 16; off; off >>= 1) part += __shfl_xor(part, off, 32);
        float sc   = part * scale;
        float mn   = fmaxf(m, sc);
        float corr = __expf(m - mn);
        float w    = __expf(sc - mn);
        s = s * corr + w;
        o = o * corr + w * vp[lane];
        m = mn;
    }
    out[((size_t)(b * Nn + i)) * (Hh * HDd) + h * HDd + lane] =
        o * __builtin_amdgcn_rcpf(s);
}

// ---------------------------------------------------------------------------
// LayerNorm over last dim (biased variance, eps=1e-5). One wave per row.
// float4-vectorized (D % 128 == 0).
// ---------------------------------------------------------------------------
__global__ __launch_bounds__(256) void layernorm_kernel(
    const float* __restrict__ in, const float* __restrict__ g,
    const float* __restrict__ bta, float* __restrict__ out, int rows, int Dd)
{
    const int wid  = threadIdx.x >> 5;
    const int lane = threadIdx.x & 31;
    int row = blockIdx.x * 8 + wid;
    if (row >= rows) return;
    const int d4c = Dd >> 2;
    const float* p = in + (size_t)row * Dd;
    float s = 0.0f, s2 = 0.0f;
    for (int d4 = lane; d4 < d4c; d4 += 32) {
        float4 v = *(const float4*)(p + d4 * 4);
        s  += v.x + v.y + v.z + v.w;
        s2 += v.x * v.x + v.y * v.y + v.z * v.z + v.w * v.w;
    }
    #pragma unroll
    for (int off = 16; off; off >>= 1) {
        s  += __shfl_xor(s,  off, 32);
        s2 += __shfl_xor(s2, off, 32);
    }
    float mean = s / (float)Dd;
    float var  = s2 / (float)Dd - mean * mean;
    float inv  = rsqrtf(var + 1e-5f);
    float* q = out + (size_t)row * Dd;
    for (int d4 = lane; d4 < d4c; d4 += 32) {
        float4 v  = *(const float4*)(p   + d4 * 4);
        float4 gg = *(const float4*)(g   + d4 * 4);
        float4 bb = *(const float4*)(bta + d4 * 4);
        float4 r;
        r.x = (v.x - mean) * inv * gg.x + bb.x;
        r.y = (v.y - mean) * inv * gg.y + bb.y;
        r.z = (v.z - mean) * inv * gg.z + bb.z;
        r.w = (v.w - mean) * inv * gg.w + bb.w;
        *(float4*)(q + d4 * 4) = r;
    }
}

// ---------------------------------------------------------------------------
// Residual add: h += y  (float4-vectorized; n % 4 == 0)
// ---------------------------------------------------------------------------
__global__ __launch_bounds__(256) void add_kernel(
    float* __restrict__ h, const float* __restrict__ y, size_t n4)
{
    size_t i = (size_t)blockIdx.x * blockDim.x + threadIdx.x;
    if (i >= n4) return;
    float4 a = *(const float4*)(h + i * 4);
    float4 b = *(const float4*)(y + i * 4);
    a.x += b.x; a.y += b.y; a.z += b.z; a.w += b.w;
    *(float4*)(h + i * 4) = a;
}

// ---------------------------------------------------------------------------
// Host-side orchestration
// ---------------------------------------------------------------------------
static inline void launch_gemm(const float* A, const float* W, const float* bias,
                               float* C, int M, int N, int K, int act,
                               hipStream_t stream)
{
    dim3 grid(N / TN, M / TM);
    gemm_bias_act_kernel<<<grid, 256, 0, stream>>>(A, W, bias, C, M, N, K, act);
}

extern "C" void kernel_launch(void* const* d_in, const int* in_sizes, int n_in,
                              void* d_out, int out_size, void* d_ws, size_t ws_size,
                              hipStream_t stream)
{
    (void)in_sizes; (void)n_in; (void)out_size; (void)ws_size;

    constexpr int Bb = 2, Nn = 1024, Dd = 1024, Hh = 16, HDd = 32;
    constexpr int DF = 4096, Ll = 2, WIN = 16, Vv = 32000;
    constexpr int NT = Bb * Nn;           // 2048 tokens

    // Inputs (setup_inputs order)
    const int*   x      = (const int*)  d_in[0];
    const float* emb    = (const float*)d_in[1];
    const float* pe     = (const float*)d_in[2];   // [1, MAXLEN, D/4]; rows of 256
    const float* pp_w1  = (const float*)d_in[3];
    const float* pp_b1  = (const float*)d_in[4];
    const float* pp_w2  = (const float*)d_in[5];
    const float* pp_b2  = (const float*)d_in[6];
    const float* pp_gw  = (const float*)d_in[7];
    const float* pp_gb  = (const float*)d_in[8];
    const float* qkv_w1 = (const float*)d_in[9];
    const float* qkv_b1 = (const float*)d_in[10];
    const float* qkv_w2 = (const float*)d_in[11];
    const float* qkv_b2 = (const float*)d_in[12];
    const float* qkv_gw = (const float*)d_in[13];
    const float* qkv_gb = (const float*)d_in[14];
    const float* out_w  = (const float*)d_in[15];
    const float* out_b  = (const float*)d_in[16];
    const float* ln1_g  = (const float*)d_in[17];
    const float* ln1_b  = (const float*)d_in[18];
    const float* kan_w1 = (const float*)d_in[19];
    const float* kan_b1 = (const float*)d_in[20];
    const float* kan_w2 = (const float*)d_in[21];
    const float* kan_b2 = (const float*)d_in[22];
    const float* kan_gw = (const float*)d_in[23];
    const float* kan_gb = (const float*)d_in[24];
    const float* ln2_g  = (const float*)d_in[25];
    const float* ln2_b  = (const float*)d_in[26];
    const float* log_w  = (const float*)d_in[27];
    const float* log_b  = (const float*)d_in[28];

    // Workspace carve-up (f32): H | T1 | T2 | QKV | O   (~88 MB total)
    float* Hbuf = (float*)d_ws;
    float* T1   = Hbuf + (size_t)NT * Dd;        // 2048*4096
    float* T2   = T1   + (size_t)NT * DF;        // 2048*4096
    float* QKV  = T2   + (size_t)NT * DF;        // 2048*1536
    float* Obuf = QKV  + (size_t)NT * 3 * Hh * HDd; // 2048*512

    // ---- positional KAN: pe[:N] (1024x256) -> 512 -> 512 -> 1024 ----
    launch_gemm(pe, pp_w1, pp_b1, T1, Nn, Dd / 2, Dd / 4, /*silu*/1, stream);
    launch_gemm(T1, pp_w2, pp_b2, T2, Nn, Dd / 2, Dd / 2, 0, stream);
    launch_gemm(T2, pp_gw, pp_gb, Obuf, Nn, Dd, Dd / 2, 0, stream);  // 1024x1024 fits in O

    // ---- embedding + positional ----
    {
        size_t tot = (size_t)NT * (Dd / 4);
        embed_kernel<<<(int)((tot + 255) / 256), 256, 0, stream>>>(
            x, emb, Obuf, Hbuf, NT, Nn, Dd);
    }

    // ---- transformer layers ----
    for (int l = 0; l < Ll; ++l) {
        const size_t wD2  = (size_t)l * Dd * Dd;
        const size_t wG   = (size_t)l * Dd * (3 * Hh * HDd);
        const size_t wO   = (size_t)l * (Hh * HDd) * Dd;
        const size_t wK1  = (size_t)l * Dd * DF;
        const size_t wK2  = (size_t)l * DF * DF;
        const size_t wKG  = (size_t)l * DF * Dd;
        const size_t bD   = (size_t)l * Dd;
        const size_t bG   = (size_t)l * (3 * Hh * HDd);
        const size_t bF   = (size_t)l * DF;

        // QKV KAN: D -> D (SiLU) -> D -> 3*H*HD
        launch_gemm(Hbuf, qkv_w1 + wD2, qkv_b1 + bD, T1, NT, Dd, Dd, 1, stream);
        launch_gemm(T1,   qkv_w2 + wD2, qkv_b2 + bD, T2, NT, Dd, Dd, 0, stream);
        launch_gemm(T2,   qkv_gw + wG,  qkv_gb + bG, QKV, NT, 3 * Hh * HDd, Dd, 0, stream);

        // attention: one wave per (b,h,i) -> B*H*N = 32768 waves, 8/block
        attn_kernel<<<(Bb * Hh * Nn + 7) / 8, 256, 0, stream>>>(
            QKV, Obuf, Bb, Nn, Hh, HDd, WIN);

        // output projection + LN1
        launch_gemm(Obuf, out_w + wO, out_b + bD, T1, NT, Dd, Hh * HDd, 0, stream);
        layernorm_kernel<<<(NT + 7) / 8, 256, 0, stream>>>(
            T1, ln1_g + bD, ln1_b + bD, T1, NT, Dd);

        // KAN feed-forward: D -> DF (SiLU) -> DF -> D, then LN2
        launch_gemm(T1, kan_w1 + wK1, kan_b1 + bF, T2, NT, DF, Dd, 1, stream);
        launch_gemm(T2, kan_w2 + wK2, kan_b2 + bF, T1, NT, DF, DF, 0, stream);
        launch_gemm(T1, kan_gw + wKG, kan_gb + bD, T2, NT, Dd, DF, 0, stream);
        layernorm_kernel<<<(NT + 7) / 8, 256, 0, stream>>>(
            T2, ln2_g + bD, ln2_b + bD, T2, NT, Dd);

        // residual
        {
            size_t tot4 = (size_t)NT * Dd / 4;
            add_kernel<<<(int)((tot4 + 255) / 256), 256, 0, stream>>>(Hbuf, T2, tot4);
        }
    }

    // ---- logits: 2048 x 32000 ----
    launch_gemm(Hbuf, log_w, log_b, (float*)d_out, NT, Vv, Dd, 0, stream);
}
